// MoveHead_42906723287273
// MI455X (gfx1250) — matrix operand order repaired and checked by reading, not compiled, not run on hardware
//
#include <hip/hip_runtime.h>
#include <hip/hip_bf16.h>

// ---------------------------------------------------------------------------
// MoveHead fused for MI455X (gfx1250, wave32, WMMA bf16).
// B=64, T=512, F=512, F2=256, F4=128, NUM_MOVES=4.  BT = 32768 slots.
// One workgroup (512 threads = 16 wave32) processes 16 slots entirely in LDS.
// ---------------------------------------------------------------------------

#define FD   512
#define FD2  256
#define FD4  128
#define NMOV 4
#define BT   32768
#define TQ   16        // slots per workgroup
#define NTHR 512       // 16 waves

typedef __attribute__((ext_vector_type(16))) __bf16        v16bf;
typedef __attribute__((ext_vector_type(8)))  float         v8f;
typedef __attribute__((ext_vector_type(8)))  unsigned int  v8u;

// Output layout in d_out (all as float, concatenated in reference return order)
// move_logits [BT*4] | move_policy [BT*4] | move_index [BT] | ar_out [BT*F] | proj [BT*F]
#define LG_OFF  ((size_t)0)
#define POL_OFF ((size_t)131072)
#define IDX_OFF ((size_t)262144)
#define ARO_OFF ((size_t)294912)
#define PRJ_OFF ((size_t)(294912 + 16777216))

// LDS carve (bytes)
#define S_AR   0                       // bf16 [16][512]   16384 B
#define S_MV   16384                   // bf16 [64][256]   32768 B
#define S_HQ   49152                   // bf16 [16][256]    8192 B
#define S_HK   57344                   // bf16 [64][256]   32768 B
#define S_Q    90112                   // f32  [16][128]    8192 B
#define S_K    98304                   // f32  [64][128]   32768 B
#define S_E    131072                  // bf16 [16][256]    8192 B
#define S_LG   139264                  // f32  [64]          256 B
#define S_RED  139520                  // f32  [64][8]      2048 B
#define S_IDX  141568                  // int  [16]           64 B
#define SMEM_BYTES 141632

// ---------------------------------------------------------------------------
// Fragment loaders.  Rowbase must be 4-byte aligned (even element offset).
// A 16x32 bf16 (ISA 7.12.2 "16-bit A-Matrix 16x32"):
//   lane l: m = l&15, h = l>>4; dword j in 0..3 -> K = 2j + 8h,
//   dword j in 4..7 -> K = 16 + 2(j-4) + 8h.
// B 32x16 bf16: lane l: n = l&15, h = l>>4; same K order as A:
//   dword j in 0..3 -> K = 2j + 8h, j in 4..7 -> K = 16 + 2(j-4) + 8h
//   (rowbase = row n of W^T, K contiguous).
// ---------------------------------------------------------------------------
__device__ inline v16bf load_afrag(const __bf16* rowbase, int h) {
  const unsigned int* p = (const unsigned int*)rowbase;
  v8u u;
#pragma unroll
  for (int j = 0; j < 8; ++j) {
    int dw = ((j < 4) ? j : (j + 4)) + 4 * h;
    u[j] = p[dw];
  }
  return __builtin_bit_cast(v16bf, u);
}

__device__ inline v16bf load_bfrag(const __bf16* rowbase, int h) {
  const unsigned int* p = (const unsigned int*)rowbase;
  v8u u;
#pragma unroll
  for (int j = 0; j < 8; ++j) u[j] = p[((j < 4) ? j : (j + 4)) + 4 * h];
  return __builtin_bit_cast(v16bf, u);
}

__device__ inline v8f wmma_bf16(v16bf a, v16bf b, v8f c) {
  return __builtin_amdgcn_wmma_f32_16x16x32_bf16(false, a, false, b, (short)0, c,
                                                 false, false);
}

__device__ inline float u01hash(unsigned int g) {
  unsigned long long x = (unsigned long long)g * 0x9E3779B97F4A7C15ull + 42ull;
  x ^= x >> 33; x *= 0xff51afd7ed558ccdULL;
  x ^= x >> 33; x *= 0xc4ceb9fe1a85ec53ULL;
  x ^= x >> 33;
  return (float)((unsigned int)(x >> 40)) * (1.0f / 16777216.0f);
}

// ---------------------------------------------------------------------------
// Weight prep: f32 [K][N] -> bf16 transposed [N][K] (K contiguous).
// ---------------------------------------------------------------------------
__global__ void movehead_wprep(const float* __restrict__ W, __bf16* __restrict__ WT,
                               int K, int N) {
  int i = blockIdx.x * blockDim.x + threadIdx.x;
  if (i >= K * N) return;
  int n = i / K;
  int k = i - n * K;
  WT[i] = (__bf16)W[(size_t)k * N + n];
}

// ---------------------------------------------------------------------------
// Fused MoveHead kernel.
// ---------------------------------------------------------------------------
__global__ __launch_bounds__(NTHR)
void movehead_fused(const int* __restrict__ ati,
                    const float* __restrict__ AR,
                    const float* __restrict__ MV,
                    const unsigned char* __restrict__ msk,
                    const float* __restrict__ bq1, const float* __restrict__ bq2,
                    const float* __restrict__ bk1, const float* __restrict__ bk2,
                    const float* __restrict__ bp,
                    const __bf16* __restrict__ WQ1T, const __bf16* __restrict__ WQ2T,
                    const __bf16* __restrict__ WK1T, const __bf16* __restrict__ WK2T,
                    const __bf16* __restrict__ WPT,
                    float* __restrict__ out) {
  extern __shared__ char smem[];
  __bf16* sAR = (__bf16*)(smem + S_AR);
  __bf16* sMV = (__bf16*)(smem + S_MV);
  __bf16* sHq = (__bf16*)(smem + S_HQ);
  __bf16* sHk = (__bf16*)(smem + S_HK);
  float*  sQ  = (float*)(smem + S_Q);
  float*  sK  = (float*)(smem + S_K);
  __bf16* sE  = (__bf16*)(smem + S_E);
  float*  sLg = (float*)(smem + S_LG);
  float*  sRd = (float*)(smem + S_RED);
  int*    sIx = (int*)(smem + S_IDX);

  const int tid  = threadIdx.x;
  const int lane = tid & 31;
  const int w    = tid >> 5;          // wave id, 0..15 (wave32)
  const int ln15 = lane & 15;
  const int h    = lane >> 4;
  const int base = blockIdx.x * TQ;   // first slot of this workgroup

  // Weight prefetch -> global_prefetch_b8 (L2-resident working set, ~786KB total)
  __builtin_prefetch(WQ1T + (size_t)(w * 16) * FD, 0, 3);
  __builtin_prefetch(WK1T + (size_t)(w * 16) * FD2, 0, 3);
  __builtin_prefetch(WPT  + (size_t)(w * 16) * FD2, 0, 3);

  // ---- Stage activations into LDS as bf16 -------------------------------
  for (int i = tid; i < TQ * FD; i += NTHR)
    sAR[i] = (__bf16)AR[(size_t)base * FD + i];
  for (int i = tid; i < TQ * NMOV * FD2; i += NTHR)
    sMV[i] = (__bf16)MV[(size_t)base * NMOV * FD2 + i];
  __syncthreads();

  // ---- q GEMM1: relu([16x512] x [512x256] + bq1) -> sHq ------------------
  {
    const int nt = w;                 // 16 waves, 16 N-tiles
    v8f c = {};
    const __bf16* brow = WQ1T + (size_t)(nt * 16 + ln15) * FD;
#pragma unroll 4
    for (int kb = 0; kb < FD / 32; ++kb) {
      v16bf a = load_afrag(sAR + ln15 * FD + kb * 32, h);
      v16bf b = load_bfrag(brow + kb * 32, h);
      c = wmma_bf16(a, b, c);
    }
    const int n = nt * 16 + ln15;
    const float bv = bq1[n];
#pragma unroll
    for (int r = 0; r < 8; ++r)
      sHq[(r + 8 * h) * FD2 + n] = (__bf16)fmaxf(c[r] + bv, 0.0f);
  }
  __syncthreads();

  // ---- q GEMM2: [16x256] x [256x128] + bq2 -> sQ (waves 0..7) ------------
  if (w < 8) {
    const int nt = w;
    v8f c = {};
    const __bf16* brow = WQ2T + (size_t)(nt * 16 + ln15) * FD2;
#pragma unroll
    for (int kb = 0; kb < FD2 / 32; ++kb) {
      v16bf a = load_afrag(sHq + ln15 * FD2 + kb * 32, h);
      v16bf b = load_bfrag(brow + kb * 32, h);
      c = wmma_bf16(a, b, c);
    }
    const int n = nt * 16 + ln15;
    const float bv = bq2[n];
#pragma unroll
    for (int r = 0; r < 8; ++r) sQ[(r + 8 * h) * FD4 + n] = c[r] + bv;
  }

  // ---- k GEMM1: relu([64x256] x [256x256] + bk1) -> sHk (4 tiles/wave) ---
  for (int t = w; t < 64; t += 16) {
    const int mt = t >> 4, nt = t & 15;
    v8f c = {};
    const __bf16* brow = WK1T + (size_t)(nt * 16 + ln15) * FD2;
#pragma unroll
    for (int kb = 0; kb < FD2 / 32; ++kb) {
      v16bf a = load_afrag(sMV + (mt * 16 + ln15) * FD2 + kb * 32, h);
      v16bf b = load_bfrag(brow + kb * 32, h);
      c = wmma_bf16(a, b, c);
    }
    const int n = nt * 16 + ln15;
    const float bv = bk1[n];
#pragma unroll
    for (int r = 0; r < 8; ++r)
      sHk[(mt * 16 + r + 8 * h) * FD2 + n] = (__bf16)fmaxf(c[r] + bv, 0.0f);
  }
  __syncthreads();

  // ---- k GEMM2: [64x256] x [256x128] + bk2 -> sK (2 tiles/wave) ----------
  for (int t = w; t < 32; t += 16) {
    const int mt = t >> 3, nt = t & 7;
    v8f c = {};
    const __bf16* brow = WK2T + (size_t)(nt * 16 + ln15) * FD2;
#pragma unroll
    for (int kb = 0; kb < FD2 / 32; ++kb) {
      v16bf a = load_afrag(sHk + (mt * 16 + ln15) * FD2 + kb * 32, h);
      v16bf b = load_bfrag(brow + kb * 32, h);
      c = wmma_bf16(a, b, c);
    }
    const int n = nt * 16 + ln15;
    const float bv = bk2[n];
#pragma unroll
    for (int r = 0; r < 8; ++r)
      sK[(mt * 16 + r + 8 * h) * FD4 + n] = c[r] + bv;
  }
  __syncthreads();

  // ---- move_logits[slot][mv] = dot(q[slot], k[slot][mv]) (len 128) -------
  {
    const int pair = tid >> 3;        // 0..63 = slot*4 + mv
    const int j    = tid & 7;
    const float* qr = sQ + (pair >> 2) * FD4;
    const float* kr = sK + pair * FD4;
    float s = 0.0f;
#pragma unroll
    for (int c = 0; c < 16; ++c) s += qr[j * 16 + c] * kr[j * 16 + c];
    sRd[pair * 8 + j] = s;
  }
  __syncthreads();
  if (tid < 64) {
    float s = 0.0f;
#pragma unroll
    for (int j = 0; j < 8; ++j) s += sRd[tid * 8 + j];
    sLg[tid] = s;
    const int g = base + (tid >> 2);
    out[LG_OFF + (size_t)g * 4 + (tid & 3)] = s;
  }
  __syncthreads();

  // ---- masked softmax + categorical sample (one thread per slot) ---------
  if (tid < TQ) {
    const int g = base + tid;
    float lg[4], le[4];
#pragma unroll
    for (int m = 0; m < 4; ++m) {
      lg[m] = sLg[tid * 4 + m];
      le[m] = msk[(size_t)g * 4 + m] ? 1.0f : 0.0f;
    }
    float lmin = fminf(fminf(lg[0], lg[1]), fminf(lg[2], lg[3]));
    float l[4];
#pragma unroll
    for (int m = 0; m < 4; ++m) l[m] = (le[m] > 0.0f) ? lg[m] : lmin;
    float lmax = fmaxf(fmaxf(l[0], l[1]), fmaxf(l[2], l[3]));
    float e[4], sum = 0.0f;
#pragma unroll
    for (int m = 0; m < 4; ++m) {
      float x = (l[m] - lmax) * le[m];
      e[m] = (le[m] > 0.0f) ? __expf(x) : 0.0f;
      sum += e[m];
    }
    const float inv = 1.0f / sum;
    const float u = u01hash((unsigned int)g);
    int idx = -1, lastLegal = 0;
    float cdf = 0.0f;
#pragma unroll
    for (int m = 0; m < 4; ++m) {
      const float p = e[m] * inv;
      out[POL_OFF + (size_t)g * 4 + m] = p;
      cdf += p;
      if (le[m] > 0.0f) lastLegal = m;
      if (idx < 0 && u < cdf && le[m] > 0.0f) idx = m;
    }
    if (idx < 0) idx = lastLegal;
    sIx[tid] = idx;
    out[IDX_OFF + (size_t)g] = (float)idx;
  }
  __syncthreads();

  // ---- gather selected move embedding ------------------------------------
  for (int i = tid; i < TQ * FD2; i += NTHR) {
    const int slot = i >> 8;          // / FD2
    const int c    = i & (FD2 - 1);
    sE[i] = sMV[(slot * NMOV + sIx[slot]) * FD2 + c];
  }
  __syncthreads();

  // ---- projection GEMM: [16x256] x [256x512] + bp, fused ar_out ----------
  for (int t = w; t < 32; t += 16) {
    const int nt = t;
    v8f c = {};
    const __bf16* brow = WPT + (size_t)(nt * 16 + ln15) * FD2;
#pragma unroll
    for (int kb = 0; kb < FD2 / 32; ++kb) {
      v16bf a = load_afrag(sE + ln15 * FD2 + kb * 32, h);
      v16bf b = load_bfrag(brow + kb * 32, h);
      c = wmma_bf16(a, b, c);
    }
    const int n = nt * 16 + ln15;
    const float bv = bp[n];
#pragma unroll
    for (int r = 0; r < 8; ++r) {
      const int m = r + 8 * h;
      const int g = base + m;
      const float p  = c[r] + bv;
      const float ar = AR[(size_t)g * FD + n];
      out[PRJ_OFF + (size_t)g * FD + n] = p;
      out[ARO_OFF + (size_t)g * FD + n] = (ati[g] == 0) ? (ar + p) : ar;
    }
  }
}

// ---------------------------------------------------------------------------
extern "C" void kernel_launch(void* const* d_in, const int* in_sizes, int n_in,
                              void* d_out, int out_size, void* d_ws, size_t ws_size,
                              hipStream_t stream) {
  const int*           ati = (const int*)d_in[0];
  const float*         AR  = (const float*)d_in[1];
  const float*         MV  = (const float*)d_in[2];
  const unsigned char* msk = (const unsigned char*)d_in[3];   // jnp bool = 1 byte
  const float* Wq1 = (const float*)d_in[4];
  const float* bq1 = (const float*)d_in[5];
  const float* Wq2 = (const float*)d_in[6];
  const float* bq2 = (const float*)d_in[7];
  const float* Wk1 = (const float*)d_in[8];
  const float* bk1 = (const float*)d_in[9];
  const float* Wk2 = (const float*)d_in[10];
  const float* bk2 = (const float*)d_in[11];
  const float* Wp  = (const float*)d_in[12];
  const float* bp  = (const float*)d_in[13];
  float* out = (float*)d_out;

  // bf16 transposed weights in workspace (786 KB total, L2-resident)
  __bf16* ws   = (__bf16*)d_ws;
  __bf16* WQ1T = ws;                  // 256 x 512
  __bf16* WQ2T = ws + 131072;         // 128 x 256
  __bf16* WK1T = ws + 163840;         // 256 x 256
  __bf16* WK2T = ws + 229376;         // 128 x 256
  __bf16* WPT  = ws + 262144;         // 512 x 256

  const int tpb = 256;
  movehead_wprep<<<(512 * 256 + tpb - 1) / tpb, tpb, 0, stream>>>(Wq1, WQ1T, 512, 256);
  movehead_wprep<<<(256 * 128 + tpb - 1) / tpb, tpb, 0, stream>>>(Wq2, WQ2T, 256, 128);
  movehead_wprep<<<(256 * 256 + tpb - 1) / tpb, tpb, 0, stream>>>(Wk1, WK1T, 256, 256);
  movehead_wprep<<<(256 * 128 + tpb - 1) / tpb, tpb, 0, stream>>>(Wk2, WK2T, 256, 128);
  movehead_wprep<<<(256 * 512 + tpb - 1) / tpb, tpb, 0, stream>>>(Wp,  WPT,  256, 512);

  movehead_fused<<<BT / TQ, NTHR, SMEM_BYTES, stream>>>(
      ati, AR, MV, msk, bq1, bq2, bk1, bk2, bp,
      WQ1T, WQ2T, WK1T, WK2T, WPT, out);
}
